// FlashAttentionModule_55009941127462
// MI455X (gfx1250) — matrix operand order repaired and checked by reading, not compiled, not run on hardware
//
#include <hip/hip_runtime.h>
#include <hip/hip_bf16.h>

#define BB 2
#define SS 2048
#define DD 2048
#define HH 16
#define HDD 128
#define PP 2048
#define LL (PP + SS)

typedef __attribute__((ext_vector_type(16))) __bf16 v16bf;
typedef __attribute__((ext_vector_type(8))) float v8f;

#if defined(__has_builtin)
#if __has_builtin(__builtin_amdgcn_global_load_async_to_lds_b128)
#define HAVE_ASYNC_LDS 1
#endif
#if __has_builtin(__builtin_amdgcn_s_wait_asynccnt)
#define HAVE_WAIT_ASYNC 1
#endif
#endif

#ifdef HAVE_ASYNC_LDS
typedef int v4i_ __attribute__((vector_size(16)));
typedef __attribute__((address_space(1))) v4i_* gp128;  // global int4* (src)
typedef __attribute__((address_space(3))) v4i_* lp128;  // LDS int4* (dst)
#endif

static __device__ inline void wait_async0() {
#ifdef HAVE_WAIT_ASYNC
    __builtin_amdgcn_s_wait_asynccnt(0);
#else
    asm volatile("s_wait_asynccnt 0x0" ::: "memory");
#endif
}

struct U16x2 { uint4 lo, hi; };

static __device__ inline v16bf make_frag(uint4 lo, uint4 hi) {
    U16x2 t{lo, hi};
    return __builtin_bit_cast(v16bf, t);
}

static __device__ inline __bf16 f2bf(float f) {
    unsigned u = __builtin_bit_cast(unsigned, f);
    unsigned r = (u + 0x7FFFu + ((u >> 16) & 1u)) >> 16;
    return __builtin_bit_cast(__bf16, (unsigned short)r);
}

static __device__ inline v8f wmma_bf16(v16bf a, v16bf b, v8f c) {
    // D = A(16x32 bf16) * B(32x16 bf16) + C(16x16 f32)
    return __builtin_amdgcn_wmma_f32_16x16x32_bf16(
        false, a, false, b, (short)0, c, false, false);
}

// ---------------------------------------------------------------------------
// past_k / past_v fp32 [B,H,P,HD] -> bf16 cache [B,H,L,HD] at offset 0
// ---------------------------------------------------------------------------
__global__ __launch_bounds__(256) void cvt_past_kernel(
    const float* __restrict__ src, __bf16* __restrict__ dst) {
    const size_t total = (size_t)BB * HH * PP * HDD;
    size_t base = ((size_t)blockIdx.x * blockDim.x + threadIdx.x) * 4;
    if (base >= total) return;
#pragma unroll
    for (int e = 0; e < 4; ++e) {
        size_t idx = base + e;
        size_t bh  = idx / ((size_t)PP * HDD);
        size_t rem = idx % ((size_t)PP * HDD);
        dst[bh * (size_t)LL * HDD + rem] = f2bf(src[idx]);
    }
}

// ---------------------------------------------------------------------------
// GEMM: out[m,n] = sum_k A[m,k] * W[n,k] + bias[n]   (x @ W^T + b)
// KIND 0: bf16 out, q head layout [B,H,S,HD]
// KIND 1: bf16 out, kv head layout [B,H,L,HD] at row offset P
// KIND 2: fp32 out, plain [M,N]; A is bf16 (attention output)
// ---------------------------------------------------------------------------
template <int KIND>
__global__ __launch_bounds__(256) void gemm_kernel(
    const float* __restrict__ A32, const __bf16* __restrict__ A16,
    const float* __restrict__ W, const float* __restrict__ bias,
    __bf16* __restrict__ out_bf, float* __restrict__ out_f) {
    const int tid  = threadIdx.x;
    const int lane = tid & 31;
    const int wave = tid >> 5;
    const int half = lane >> 4;
    const int l15  = lane & 15;
    const int n0 = blockIdx.x * 128;
    const int m0 = blockIdx.y * 128;

    __shared__ __bf16 As[128][40];  // 128 x 32 (+8 pad), row-major [m][k]
    __shared__ __bf16 Bs[128][40];  // 128 x 32 (+8 pad), [n][k] (= W row)

    const int mw = (wave & 3) * 32;   // 4 M-groups of 32
    const int nw = (wave >> 2) * 64;  // 2 N-groups of 64

    v8f acc[2][4];
#pragma unroll
    for (int i = 0; i < 2; ++i)
#pragma unroll
        for (int j = 0; j < 4; ++j) acc[i][j] = v8f{};

    for (int k0 = 0; k0 < DD; k0 += 32) {
        // ---- stage A tile ----
        if (KIND == 2) {
#pragma unroll
            for (int it = 0; it < 2; ++it) {  // 512 uint4 chunks
                int lin = tid + it * 256;
                int row = lin >> 2;
                int c   = lin & 3;
                const __bf16* g = A16 + (size_t)(m0 + row) * DD + k0 + c * 8;
#ifdef HAVE_ASYNC_LDS
                __builtin_amdgcn_global_load_async_to_lds_b128(
                    (gp128)g, (lp128)&As[row][c * 8], 0, 0);
#else
                uint4 v = *reinterpret_cast<const uint4*>(g);
                *reinterpret_cast<uint4*>(&As[row][c * 8]) = v;
#endif
            }
        } else {
#pragma unroll
            for (int it = 0; it < 4; ++it) {  // 1024 float4 chunks
                int lin = tid + it * 256;
                int row = lin >> 3;
                int c   = lin & 7;
                float4 f = *reinterpret_cast<const float4*>(
                    A32 + (size_t)(m0 + row) * DD + k0 + c * 4);
                As[row][c * 4 + 0] = f2bf(f.x);
                As[row][c * 4 + 1] = f2bf(f.y);
                As[row][c * 4 + 2] = f2bf(f.z);
                As[row][c * 4 + 3] = f2bf(f.w);
            }
        }
        // ---- stage B tile (rows of W, since B = W^T columns) ----
#pragma unroll
        for (int it = 0; it < 4; ++it) {
            int lin = tid + it * 256;
            int row = lin >> 3;
            int c   = lin & 7;
            float4 f = *reinterpret_cast<const float4*>(
                W + (size_t)(n0 + row) * DD + k0 + c * 4);
            Bs[row][c * 4 + 0] = f2bf(f.x);
            Bs[row][c * 4 + 1] = f2bf(f.y);
            Bs[row][c * 4 + 2] = f2bf(f.z);
            Bs[row][c * 4 + 3] = f2bf(f.w);
        }
#ifdef HAVE_ASYNC_LDS
        if (KIND == 2) wait_async0();
#endif
        __syncthreads();

        // ---- fragments + WMMA ----
        v16bf afr[2], bfr[4];
#pragma unroll
        for (int i = 0; i < 2; ++i) {
            int r  = mw + i * 16 + l15;
            int c0 = half * 8;  // A layout: lanes 0-15 K 0-7/16-23; 16-31 K 8-15/24-31
            uint4 lo = *reinterpret_cast<const uint4*>(&As[r][c0]);
            uint4 hi = *reinterpret_cast<const uint4*>(&As[r][c0 + 16]);
            afr[i] = make_frag(lo, hi);
        }
#pragma unroll
        for (int j = 0; j < 4; ++j) {
            int n  = nw + j * 16 + l15;
            int c0 = half * 8;  // B layout: lanes 0-15 K 0-7/16-23; 16-31 K 8-15/24-31
            uint4 lo = *reinterpret_cast<const uint4*>(&Bs[n][c0]);
            uint4 hi = *reinterpret_cast<const uint4*>(&Bs[n][c0 + 16]);
            bfr[j] = make_frag(lo, hi);
        }
#pragma unroll
        for (int i = 0; i < 2; ++i)
#pragma unroll
            for (int j = 0; j < 4; ++j)
                acc[i][j] = wmma_bf16(afr[i], bfr[j], acc[i][j]);
        __syncthreads();
    }

    // ---- epilogue: bias + store ----
#pragma unroll
    for (int i = 0; i < 2; ++i) {
#pragma unroll
        for (int j = 0; j < 4; ++j) {
            int ncol = n0 + nw + j * 16 + l15;
            float bv = bias[ncol];
#pragma unroll
            for (int r = 0; r < 8; ++r) {
                int mrow  = m0 + mw + i * 16 + r + half * 8;  // C layout
                float val = acc[i][j][r] + bv;
                if (KIND == 2) {
                    out_f[(size_t)mrow * DD + ncol] = val;
                } else {
                    int b  = mrow >> 11;
                    int s  = mrow & (SS - 1);
                    int h  = ncol >> 7;
                    int hd = ncol & (HDD - 1);
                    size_t idx;
                    if (KIND == 0)
                        idx = ((size_t)(b * HH + h) * SS + s) * HDD + hd;
                    else
                        idx = ((size_t)(b * HH + h) * LL + PP + s) * HDD + hd;
                    out_bf[idx] = f2bf(val);
                }
            }
        }
    }
}

// ---------------------------------------------------------------------------
// Flash attention: q [B,H,S,HD] bf16, k/v cache [B,H,L,HD] bf16.
// Causal offset mask: qpos = P + s, keep k <= qpos. Softmax online.
// Grid (S/128, H, B); 256 threads = 8 waves; each wave = 16 q rows.
// ---------------------------------------------------------------------------
__global__ __launch_bounds__(256) void attn_kernel(
    const __bf16* __restrict__ qb, const __bf16* __restrict__ kb,
    const __bf16* __restrict__ vb, __bf16* __restrict__ ob) {
    const int tid  = threadIdx.x;
    const int lane = tid & 31;
    const int wave = tid >> 5;
    const int half = lane >> 4;
    const int l15  = lane & 15;
    const int blkq = blockIdx.x;
    const int h    = blockIdx.y;
    const int b    = blockIdx.z;
    const int bh   = b * HH + h;
    const int qt   = blkq * 128 + wave * 16;

    __shared__ __bf16 Ks[32][136];     // [keypos][hd], padded
    __shared__ __bf16 Vs[128][40];     // transposed: [hd][keypos], padded
    __shared__ __bf16 Pw[8][16][32];   // per-wave P scratch [M][k]

    // Q fragments, loaded directly from global in A-layout
    v16bf qf[4];
    {
        const __bf16* qrow = qb + ((size_t)bh * SS + qt + l15) * HDD;
#pragma unroll
        for (int j = 0; j < 4; ++j) {
            int c0 = j * 32 + half * 8;
            uint4 lo = *reinterpret_cast<const uint4*>(qrow + c0);
            uint4 hi = *reinterpret_cast<const uint4*>(qrow + c0 + 16);
            qf[j] = make_frag(lo, hi);
        }
    }

    float mrun[8], lrun[8];
    v8f o[8];
#pragma unroll
    for (int r = 0; r < 8; ++r) { mrun[r] = -1e30f; lrun[r] = 0.f; }
#pragma unroll
    for (int t = 0; t < 8; ++t) o[t] = v8f{};

    const float scale = 0.08838834764831845f;  // 1/sqrt(128)
    const int nblk = (PP + (blkq + 1) * 128) / 32;
    const __bf16* kbase = kb + (size_t)bh * LL * HDD;
    const __bf16* vbase = vb + (size_t)bh * LL * HDD;

    for (int blk = 0; blk < nblk; ++blk) {
        const int kb0 = blk * 32;
        // ---- cooperative K stage: 32x128 bf16 = 512 x b128 ----
#pragma unroll
        for (int it = 0; it < 2; ++it) {
            int lin = tid + it * 256;
            int row = lin >> 4;
            int c   = lin & 15;
            const __bf16* g = kbase + (size_t)(kb0 + row) * HDD + c * 8;
#ifdef HAVE_ASYNC_LDS
            __builtin_amdgcn_global_load_async_to_lds_b128(
                (gp128)g, (lp128)&Ks[row][c * 8], 0, 0);
#else
            uint4 v = *reinterpret_cast<const uint4*>(g);
            *reinterpret_cast<uint4*>(&Ks[row][c * 8]) = v;
#endif
        }
        // ---- cooperative V stage, transposed into [hd][pos] ----
        {
            int pp = tid >> 4;  // key-position pair 0..15
            int hc = tid & 15;  // hd chunk of 8
            const __bf16* v0p = vbase + (size_t)(kb0 + pp * 2) * HDD + hc * 8;
            uint4 a = *reinterpret_cast<const uint4*>(v0p);
            uint4 bq4 = *reinterpret_cast<const uint4*>(v0p + HDD);
            unsigned aw[4] = {a.x, a.y, a.z, a.w};
            unsigned bw[4] = {bq4.x, bq4.y, bq4.z, bq4.w};
#pragma unroll
            for (int i = 0; i < 4; ++i) {
                unsigned lo = (aw[i] & 0xFFFFu) | (bw[i] << 16);
                unsigned hi = (aw[i] >> 16) | (bw[i] & 0xFFFF0000u);
                int hd = hc * 8 + i * 2;
                *reinterpret_cast<unsigned*>(&Vs[hd][pp * 2])     = lo;
                *reinterpret_cast<unsigned*>(&Vs[hd + 1][pp * 2]) = hi;
            }
        }
        // prefetch next K/V block while this one is consumed
        if (blk + 1 < nblk) {
            const __bf16* nk = kbase + (size_t)(kb0 + 32) * HDD;
            const __bf16* nv = vbase + (size_t)(kb0 + 32) * HDD;
            __builtin_prefetch(nk + tid * 32, 0, 0);
            __builtin_prefetch(nv + tid * 32, 0, 0);
        }
#ifdef HAVE_ASYNC_LDS
        wait_async0();
#endif
        __syncthreads();

        // ---- scores: S(16x32) = Q(16x128) * K^T(128x32) ----
        v8f s0 = v8f{}, s1 = v8f{};
#pragma unroll
        for (int j = 0; j < 4; ++j) {
            int c0 = j * 32 + half * 8;
            uint4 lo = *reinterpret_cast<const uint4*>(&Ks[l15][c0]);
            uint4 hi = *reinterpret_cast<const uint4*>(&Ks[l15][c0 + 16]);
            v16bf bk0 = make_frag(lo, hi);
            lo = *reinterpret_cast<const uint4*>(&Ks[16 + l15][c0]);
            hi = *reinterpret_cast<const uint4*>(&Ks[16 + l15][c0 + 16]);
            v16bf bk1 = make_frag(lo, hi);
            s0 = wmma_bf16(qf[j], bk0, s0);
            s1 = wmma_bf16(qf[j], bk1, s1);
        }

        // ---- online softmax; mask only in diagonal blocks ----
        auto softmax_update = [&](bool domask) __attribute__((always_inline)) {
#pragma unroll
            for (int r = 0; r < 8; ++r) {
                float a0 = s0[r] * scale;
                float a1 = s1[r] * scale;
                if (domask) {
                    int qpos = PP + qt + r + half * 8;
                    if (kb0 + l15 > qpos)      a0 = -1e30f;
                    if (kb0 + 16 + l15 > qpos) a1 = -1e30f;
                }
                float rm = fmaxf(a0, a1);
#pragma unroll
                for (int off = 1; off < 16; off <<= 1)
                    rm = fmaxf(rm, __shfl_xor(rm, off, 32));
                float mn    = fmaxf(mrun[r], rm);
                float alpha = __expf(mrun[r] - mn);
                float p0    = __expf(a0 - mn);
                float p1    = __expf(a1 - mn);
                float rs    = p0 + p1;
#pragma unroll
                for (int off = 1; off < 16; off <<= 1)
                    rs += __shfl_xor(rs, off, 32);
                lrun[r] = lrun[r] * alpha + rs;
                mrun[r] = mn;
#pragma unroll
                for (int t = 0; t < 8; ++t) o[t][r] *= alpha;
                Pw[wave][r + half * 8][l15]      = f2bf(p0);
                Pw[wave][r + half * 8][16 + l15] = f2bf(p1);
            }
        };
        if (kb0 + 31 > PP + qt) softmax_update(true);
        else                    softmax_update(false);
        asm volatile("s_wait_dscnt 0x0" ::: "memory");  // same-wave LDS turnaround

        // ---- P fragment (A layout, 16x32 over key positions) ----
        v16bf pf;
        {
            const __bf16* prow = &Pw[wave][l15][0];
            int c0 = half * 8;
            uint4 lo = *reinterpret_cast<const uint4*>(prow + c0);
            uint4 hi = *reinterpret_cast<const uint4*>(prow + c0 + 16);
            pf = make_frag(lo, hi);
        }
        // ---- O(16x128) += P(16x32) * V(32x128) ----
#pragma unroll
        for (int t = 0; t < 8; ++t) {
            int c0 = half * 8;
            uint4 lo = *reinterpret_cast<const uint4*>(&Vs[t * 16 + l15][c0]);
            uint4 hi = *reinterpret_cast<const uint4*>(&Vs[t * 16 + l15][c0 + 16]);
            v16bf vf = make_frag(lo, hi);
            o[t] = wmma_bf16(pf, vf, o[t]);
        }
        __syncthreads();
    }

    // ---- normalize + store bf16 [B,S,D] ----
#pragma unroll
    for (int r = 0; r < 8; ++r) {
        float inv = 1.0f / lrun[r];
        int srow  = qt + r + half * 8;
        __bf16* orow = ob + ((size_t)b * SS + srow) * DD + h * HDD;
#pragma unroll
        for (int t = 0; t < 8; ++t)
            orow[t * 16 + l15] = f2bf(o[t][r] * inv);
    }
}

// ---------------------------------------------------------------------------
extern "C" void kernel_launch(void* const* d_in, const int* in_sizes, int n_in,
                              void* d_out, int out_size, void* d_ws, size_t ws_size,
                              hipStream_t stream) {
    const float* x      = (const float*)d_in[0];
    const float* past_k = (const float*)d_in[1];
    const float* past_v = (const float*)d_in[2];
    const float* Wq = (const float*)d_in[3];
    const float* bq = (const float*)d_in[4];
    const float* Wk = (const float*)d_in[5];
    const float* bk = (const float*)d_in[6];
    const float* Wv = (const float*)d_in[7];
    const float* bv = (const float*)d_in[8];
    const float* Wo = (const float*)d_in[9];
    const float* bo = (const float*)d_in[10];
    float* out = (float*)d_out;

    const size_t QB = (size_t)BB * HH * SS * HDD * sizeof(__bf16);  // 16 MB
    const size_t KB = (size_t)BB * HH * LL * HDD * sizeof(__bf16);  // 32 MB
    char* ws = (char*)d_ws;
    __bf16* qb_ = (__bf16*)(ws);
    __bf16* kb_ = (__bf16*)(ws + QB);
    __bf16* vb_ = (__bf16*)(ws + QB + KB);
    __bf16* ob_ = (__bf16*)(ws + QB + 2 * KB);

    // 1) past KV fp32 -> bf16 cache (rows [0,P))
    {
        const size_t total = (size_t)BB * HH * PP * HDD;
        int blocks = (int)((total / 4 + 255) / 256);
        cvt_past_kernel<<<blocks, 256, 0, stream>>>(past_k, kb_);
        cvt_past_kernel<<<blocks, 256, 0, stream>>>(past_v, vb_);
    }

    // 2) QKV projections (K/V new rows written at cache offset P)
    dim3 gg(DD / 128, (BB * SS) / 128);
    gemm_kernel<0><<<gg, 256, 0, stream>>>(x, nullptr, Wq, bq, qb_, nullptr);
    gemm_kernel<1><<<gg, 256, 0, stream>>>(x, nullptr, Wk, bk, kb_, nullptr);
    gemm_kernel<1><<<gg, 256, 0, stream>>>(x, nullptr, Wv, bv, vb_, nullptr);

    // 3) causal flash attention
    attn_kernel<<<dim3(SS / 128, HH, BB), 256, 0, stream>>>(qb_, kb_, vb_, ob_);

    // 4) output projection -> fp32 d_out
    gemm_kernel<2><<<gg, 256, 0, stream>>>(nullptr, ob_, Wo, bo, nullptr, out);
}
